// LIFLayer_58841051955588
// MI455X (gfx1250) — matrix operand order repaired and checked by reading, not compiled, not run on hardware
//
#include <hip/hip_runtime.h>

typedef __attribute__((ext_vector_type(16))) __bf16 v16bf;
typedef __attribute__((ext_vector_type(2)))  __bf16 v2bf;
typedef __attribute__((ext_vector_type(8)))  float  v8f;

namespace {
constexpr int T   = 32;
constexpr int BAT = 256;   // batch
constexpr int CI  = 4096;  // input features
constexpr int CO  = 4096;  // output features

constexpr int BM  = 64;    // batch rows per block
constexpr int BN  = 128;   // output cols per block
constexpr int KC  = 32;    // K chunk (one WMMA K step)
constexpr int LDA = KC + 8;  // padded LDS row pitch (bf16 elems) -> 80B
constexpr int LDB = KC + 8;
constexpr int NKC = CI / KC; // 128 chunks (even, required by 2x unroll)
}

// f32 pair -> packed bf16x2 via hardware packed convert
__device__ __forceinline__ unsigned pack_bf16x2(float lo, float hi) {
#if __has_builtin(__builtin_amdgcn_cvt_pk_bf16_f32)
  v2bf p = __builtin_amdgcn_cvt_pk_bf16_f32(lo, hi);
  return __builtin_bit_cast(unsigned, p);
#else
  v2bf p;
  p[0] = (__bf16)lo;
  p[1] = (__bf16)hi;
  return __builtin_bit_cast(unsigned, p);
#endif
}

// Workgroup barrier that drains ONLY DScnt (LDS visibility), leaving the
// global-load prefetch in flight across the rendezvous. __syncthreads()
// would emit s_wait_loadcnt_dscnt 0x0 and stall on our own pipeline.
__device__ __forceinline__ void lds_barrier() {
  asm volatile("s_wait_dscnt 0x0\n\t"
               "s_barrier_signal -1\n\t"
               "s_barrier_wait -1" ::: "memory");
}

__global__ __launch_bounds__(256)
void lif_wmma_kernel(const float* __restrict__ x, const float* __restrict__ W,
                     const float* __restrict__ bias,
                     const float* __restrict__ decay,
                     const float* __restrict__ thresh,
                     float* __restrict__ out)
{
  // ping-pong double buffers (~30 KB of the 320 KB WGP pool)
  __shared__ __align__(16) unsigned short ldsA[2][BM * LDA];
  __shared__ __align__(16) unsigned short ldsB[2][BN * LDB];

  const int tid  = threadIdx.x;
  const int lane = tid & 31;
  const int wv   = tid >> 5;   // wave 0..7
  const int wm   = wv >> 2;    // wave row 0..1  (32 batch rows each)
  const int wn   = wv & 3;     // wave col 0..3  (32 out cols each)
  const int lr   = lane & 15;  // row/col within 16x16 tile
  const int hh   = lane >> 4;  // half selector

  const int n0 = blockIdx.x * BN;
  const int m0 = blockIdx.y * BM;

  const float dcy = decay[0];
  const float th  = thresh[0];

  int   colg[2];
  float bv[2];
#pragma unroll
  for (int ni = 0; ni < 2; ++ni) {
    colg[ni] = n0 + wn * 32 + ni * 16 + lr;
    bv[ni]   = bias[colg[ni]];
  }

  // persistent membrane state: 2x2 subtiles of 16x16, f32 C/D layout
  v8f mem[2][2] = {};

  // cooperative loader mapping: 8 float4 per row-of-32-K
  const int ldrow = tid >> 3;        // 0..31
  const int ldk   = (tid & 7) * 4;   // K offset within chunk

  float4 pa[2], pb[4];               // in-flight chunk registers
  v8f acc[2][2];

  // ---- helpers -------------------------------------------------------
  auto loadchunk = [&](const float* xt, int k0) {
#pragma unroll
    for (int i = 0; i < 2; ++i)
      pa[i] = *(const float4*)(xt + (size_t)(ldrow + i * 32) * CI + k0 + ldk);
#pragma unroll
    for (int i = 0; i < 4; ++i)
      pb[i] = *(const float4*)(W + (size_t)(n0 + ldrow + i * 32) * CI + k0 + ldk);
  };

  auto stage = [&](unsigned short* dA, unsigned short* dB) {
#pragma unroll
    for (int i = 0; i < 2; ++i) {
      uint2 w2;
      w2.x = pack_bf16x2(pa[i].x, pa[i].y);
      w2.y = pack_bf16x2(pa[i].z, pa[i].w);
      *(uint2*)&dA[(ldrow + i * 32) * LDA + ldk] = w2;
    }
#pragma unroll
    for (int i = 0; i < 4; ++i) {
      uint2 w2;
      w2.x = pack_bf16x2(pb[i].x, pb[i].y);
      w2.y = pack_bf16x2(pb[i].z, pb[i].w);
      *(uint2*)&dB[(ldrow + i * 32) * LDB + ldk] = w2;
    }
  };

  auto compute = [&](const unsigned short* sA, const unsigned short* sB) {
    union FragU { v16bf v; uint4 q[2]; };
    FragU af[2], bfr[2];
#pragma unroll
    for (int mi = 0; mi < 2; ++mi) {
      // A 16x32: lane = M row; 16 contiguous K at hh*16
      const unsigned short* p = &sA[(wm * 32 + mi * 16 + lr) * LDA];
      af[mi].q[0] = *(const uint4*)(p + hh * 16);
      af[mi].q[1] = *(const uint4*)(p + hh * 16 + 8);
    }
#pragma unroll
    for (int ni = 0; ni < 2; ++ni) {
      // B 32x16: lane = N col; 16 contiguous K at hh*16
      const unsigned short* p = &sB[(wn * 32 + ni * 16 + lr) * LDB + hh * 16];
      bfr[ni].q[0] = *(const uint4*)(p);
      bfr[ni].q[1] = *(const uint4*)(p + 8);
    }
#pragma unroll
    for (int mi = 0; mi < 2; ++mi)
#pragma unroll
      for (int ni = 0; ni < 2; ++ni)
        acc[mi][ni] = __builtin_amdgcn_wmma_f32_16x16x32_bf16(
            false, af[mi].v, false, bfr[ni].v, (short)0, acc[mi][ni],
            false, false);
  };
  // --------------------------------------------------------------------

#pragma unroll 1
  for (int t = 0; t < T; ++t) {
#pragma unroll
    for (int mi = 0; mi < 2; ++mi)
#pragma unroll
      for (int ni = 0; ni < 2; ++ni)
        acc[mi][ni] = v8f{};

    const float* xt = x + ((size_t)t * BAT + m0) * CI;

    // prologue: stage chunk 0 into buf0, put chunk 1 in flight
    loadchunk(xt, 0);
    stage(ldsA[0], ldsB[0]);
    loadchunk(xt, KC);
    lds_barrier();

#pragma unroll 1
    for (int kc2 = 0; kc2 < NKC; kc2 += 2) {
      // even chunk from buf0; stage odd chunk into buf1 while WMMAs drain
      compute(ldsA[0], ldsB[0]);
      stage(ldsA[1], ldsB[1]);                 // chunk kc2+1 (always exists)
      if (kc2 + 2 < NKC) loadchunk(xt, (kc2 + 2) * KC);
      lds_barrier();

      // odd chunk from buf1; stage next even chunk into buf0
      compute(ldsA[1], ldsB[1]);
      if (kc2 + 2 < NKC) {
        stage(ldsA[0], ldsB[0]);               // chunk kc2+2
        if (kc2 + 3 < NKC) loadchunk(xt, (kc2 + 3) * KC);
      }
      lds_barrier();
    }

    // LIF update + spike store (C/D layout: reg r -> M = r + 8*hh, N = lr)
#pragma unroll
    for (int mi = 0; mi < 2; ++mi) {
      int rowbase = m0 + wm * 32 + mi * 16 + hh * 8;
#pragma unroll
      for (int ni = 0; ni < 2; ++ni) {
        float* op = out + ((size_t)t * BAT + rowbase) * CO + colg[ni];
#pragma unroll
        for (int r = 0; r < 8; ++r) {
          float m = mem[mi][ni][r];
          m = dcy * m + (acc[mi][ni][r] + bv[ni]);   // leaky integrate
          float s = (m - th) > 0.0f ? 1.0f : 0.0f;   // fire
          op[(size_t)r * CO] = s;
          mem[mi][ni][r] = m - s * th;               // reset by subtraction
        }
      }
    }
  }
}

extern "C" void kernel_launch(void* const* d_in, const int* in_sizes, int n_in,
                              void* d_out, int out_size, void* d_ws, size_t ws_size,
                              hipStream_t stream) {
  (void)in_sizes; (void)n_in; (void)d_ws; (void)ws_size; (void)out_size;
  const float* x      = (const float*)d_in[0];
  const float* W      = (const float*)d_in[1];
  const float* bias   = (const float*)d_in[2];
  const float* decay  = (const float*)d_in[3];
  const float* thresh = (const float*)d_in[4];
  float* out = (float*)d_out;

  dim3 grid(CO / BN, BAT / BM);   // (32, 4) = 128 blocks
  lif_wmma_kernel<<<grid, dim3(256), 0, stream>>>(x, W, bias, decay, thresh, out);
}
